// StandardAttention_5927054868612
// MI455X (gfx1250) — matrix operand order repaired and checked by reading, not compiled, not run on hardware
//
#include <hip/hip_runtime.h>

// ---------------------------------------------------------------------------
// Types & fragment helpers (CDNA5 wave32 WMMA, bf16 -> f32 accumulate)
// ---------------------------------------------------------------------------
typedef __attribute__((ext_vector_type(16))) __bf16        v16bf;
typedef __attribute__((ext_vector_type(8)))  float         v8f;
typedef __attribute__((ext_vector_type(8)))  unsigned short u16x8;
typedef __attribute__((ext_vector_type(16))) unsigned short u16x16;

// A-matrix fragment (16x32 bf16): per lane two contiguous 8-halfword chunks.
union FragA { u16x8 h[2]; v16bf v; };
// B-matrix fragment (32x16 bf16): per lane one contiguous 16-halfword chunk.
union FragB { u16x16 u; v16bf v; };

__device__ __forceinline__ unsigned short f2bf(float f) {
  unsigned int u = __float_as_uint(f);
  return (unsigned short)((u + 0x7FFFu + ((u >> 16) & 1u)) >> 16);  // RNE
}

__device__ __forceinline__ v8f wmma_bf16(v16bf a, v16bf b, v8f c) {
  // (neg_a, A, neg_b, B, c_mod, C, reuse_a, reuse_b)
  return __builtin_amdgcn_wmma_f32_16x16x32_bf16(false, a, false, b,
                                                 (short)0, c, false, false);
}

// XOR-swizzle within 32 lanes (group-of-32 mode: and=0x1F, or=0, xor=m).
#define SWZ_XOR(v, m)                                                         \
  __builtin_bit_cast(float, __builtin_amdgcn_ds_swizzle(                      \
      __builtin_bit_cast(int, (float)(v)), (((m) << 10) | 0x1F)))

__device__ __forceinline__ float redmax16(float v) {
  v = fmaxf(v, SWZ_XOR(v, 1));
  v = fmaxf(v, SWZ_XOR(v, 2));
  v = fmaxf(v, SWZ_XOR(v, 4));
  v = fmaxf(v, SWZ_XOR(v, 8));
  return v;
}
__device__ __forceinline__ float redsum16(float v) {
  v += SWZ_XOR(v, 1);
  v += SWZ_XOR(v, 2);
  v += SWZ_XOR(v, 4);
  v += SWZ_XOR(v, 8);
  return v;
}

// Problem constants
#define BATCH  2
#define SEQ    4096
#define DIM    512
#define HEADS  8
#define DH     64
#define ROWS   (BATCH * SEQ)   // 8192
#define E_QKV  (3 * DIM)       // 1536
#define SOFTMAX_SCALE 0.125f   // 64^-0.5

// ---------------------------------------------------------------------------
// fp32 -> bf16 conversion
// ---------------------------------------------------------------------------
__global__ void cvt_bf16(const float* __restrict__ src,
                         unsigned short* __restrict__ dst, int n) {
  int i = blockIdx.x * blockDim.x + threadIdx.x;
  if (i < n) dst[i] = f2bf(src[i]);
}

// ---------------------------------------------------------------------------
// QKV projection: qkv = x @ w_qkv^T   (M=8192, N=1536, K=512)
// Epilogue scatters into Q [b,h,n,d], K [b,h,n,d], V^T [b,h,d,n] (bf16).
// Block: 128 threads = 4 waves; wave computes a 16x64 output tile.
// ---------------------------------------------------------------------------
__global__ void __launch_bounds__(128)
qkv_gemm(const unsigned short* __restrict__ xb,
         const unsigned short* __restrict__ wb,
         unsigned short* __restrict__ qb,
         unsigned short* __restrict__ kb,
         unsigned short* __restrict__ vtb) {
  const int lane = threadIdx.x & 31;
  const int wave = threadIdx.x >> 5;
  const int l15  = lane & 15;
  const int hi   = lane >> 4;
  const int m0   = blockIdx.x * 16;
  const int e0   = blockIdx.y * 256 + wave * 64;

  v8f acc[4] = {};
  const unsigned short* arow = xb + (size_t)(m0 + l15) * DIM;

  for (int kk = 0; kk < DIM; kk += 32) {
    FragA a;
    a.h[0] = *(const u16x8*)(arow + kk + hi * 16);
    a.h[1] = *(const u16x8*)(arow + kk + hi * 16 + 8);
#pragma unroll
    for (int t = 0; t < 4; ++t) {
      FragB bf;
      bf.u = *(const u16x16*)(wb + (size_t)(e0 + t * 16 + l15) * DIM + kk + hi * 16);
      acc[t] = wmma_bf16(a.v, bf.v, acc[t]);
    }
  }

#pragma unroll
  for (int t = 0; t < 4; ++t) {
#pragma unroll
    for (int r = 0; r < 8; ++r) {
      const int row = m0 + r + hi * 8;       // global token row
      const int bi  = row >> 12;             // batch
      const int n   = row & 4095;            // token in batch
      const int e   = e0 + t * 16 + l15;     // output feature
      const unsigned short bv = f2bf(acc[t][r]);
      if (e < DIM) {                         // Q: [b,h,n,d]
        const int h = e >> 6, d = e & 63;
        qb[(((size_t)(bi * HEADS + h)) * SEQ + n) * DH + d] = bv;
      } else if (e < 2 * DIM) {              // K: [b,h,n,d]
        const int ee = e - DIM;
        const int h = ee >> 6, d = ee & 63;
        kb[(((size_t)(bi * HEADS + h)) * SEQ + n) * DH + d] = bv;
      } else {                               // V transposed: [b,h,d,n]
        const int ee = e - 2 * DIM;
        const int h = ee >> 6, d = ee & 63;
        vtb[(((size_t)(bi * HEADS + h)) * DH + d) * SEQ + n] = bv;
      }
    }
  }
}

// ---------------------------------------------------------------------------
// Flash attention: per (b,h,64-query block). 4 waves x 16 query rows.
// Streams 64-key K/V blocks through LDS; online softmax in f32.
// ---------------------------------------------------------------------------
__global__ void __launch_bounds__(128)
flash_attn(const unsigned short* __restrict__ qg,
           const unsigned short* __restrict__ kg,
           const unsigned short* __restrict__ vtg,
           unsigned short* __restrict__ og) {
  __shared__ __align__(32) unsigned short kt[64 * 64];      // K block [j][d]
  __shared__ __align__(32) unsigned short vt[64 * 64];      // V block [d][j]
  __shared__ __align__(32) unsigned short pb[4][16 * 64];   // per-wave P scratch

  const int tid  = threadIdx.x;
  const int lane = tid & 31;
  const int wave = tid >> 5;
  const int l15  = lane & 15;
  const int hi   = lane >> 4;
  const int h    = blockIdx.y;
  const int b    = blockIdx.z;
  const int bh   = b * HEADS + h;
  const int q0   = blockIdx.x * 64 + wave * 16;

  const unsigned short* qbase = qg  + (size_t)bh * SEQ * DH;
  const unsigned short* kbase = kg  + (size_t)bh * SEQ * DH;
  const unsigned short* vbase = vtg + (size_t)bh * DH * SEQ;

  // Q A-fragments (16 rows x 64 d) held in registers for the whole kernel.
  FragA qa[2];
  {
    const unsigned short* qr = qbase + (size_t)(q0 + l15) * DH;
#pragma unroll
    for (int f = 0; f < 2; ++f) {
      qa[f].h[0] = *(const u16x8*)(qr + f * 32 + hi * 16);
      qa[f].h[1] = *(const u16x8*)(qr + f * 32 + hi * 16 + 8);
    }
  }

  v8f o[4] = {};
  float m_i[8], l_i[8];
#pragma unroll
  for (int r = 0; r < 8; ++r) { m_i[r] = -__builtin_inff(); l_i[r] = 0.0f; }

  for (int jb = 0; jb < SEQ; jb += 64) {
    __syncthreads();  // previous iteration's LDS reads done
    // Cooperative load: K block (contiguous 8 KB) + V^T block (64 strided rows).
#pragma unroll
    for (int i = 0; i < 4; ++i) {
      const int idx = tid + i * 128;                     // 0..511 (8 hw each)
      *(u16x8*)(kt + idx * 8) = *(const u16x8*)(kbase + (size_t)jb * DH + idx * 8);
      const int d = idx >> 3, c = idx & 7;
      *(u16x8*)(vt + d * 64 + c * 8) =
          *(const u16x8*)(vbase + (size_t)d * SEQ + jb + c * 8);
    }
    if (jb + 64 < SEQ) {  // prefetch next block (global_prefetch)
      __builtin_prefetch(kbase + (size_t)(jb + 64) * DH + tid * 64, 0, 3);
      __builtin_prefetch(vbase + (size_t)(tid & 63) * SEQ + jb + 64, 0, 3);
    }
    __syncthreads();

    // S = Q * K^T  (16 x 64 per wave)
    v8f s[4] = {};
#pragma unroll
    for (int t = 0; t < 4; ++t) {
#pragma unroll
      for (int f = 0; f < 2; ++f) {
        FragB kf;
        kf.u = *(const u16x16*)(kt + (t * 16 + l15) * 64 + f * 32 + hi * 16);
        s[t] = wmma_bf16(qa[f].v, kf.v, s[t]);
      }
    }

    // Online softmax (row = r + 8*hi; row values live across 16 lanes).
    float alpha[8];
#pragma unroll
    for (int r = 0; r < 8; ++r) {
      float mx = fmaxf(fmaxf(s[0][r], s[1][r]), fmaxf(s[2][r], s[3][r]));
      mx = redmax16(mx) * SOFTMAX_SCALE;
      const float mn = fmaxf(m_i[r], mx);
      alpha[r] = __expf(m_i[r] - mn);
      m_i[r] = mn;
    }
    float rs[8] = {0, 0, 0, 0, 0, 0, 0, 0};
#pragma unroll
    for (int t = 0; t < 4; ++t) {
#pragma unroll
      for (int r = 0; r < 8; ++r) {
        const float p = __expf(s[t][r] * SOFTMAX_SCALE - m_i[r]);
        rs[r] += p;
        pb[wave][(r + hi * 8) * 64 + t * 16 + l15] = f2bf(p);  // C -> LDS
      }
    }
#pragma unroll
    for (int r = 0; r < 8; ++r) l_i[r] = l_i[r] * alpha[r] + redsum16(rs[r]);
#pragma unroll
    for (int t = 0; t < 4; ++t)
#pragma unroll
      for (int r = 0; r < 8; ++r) o[t][r] *= alpha[r];

    // Per-wave LDS round-trip: make P stores visible before A-frag reloads.
    asm volatile("s_wait_dscnt 0" ::: "memory");

    // O += P * V
#pragma unroll
    for (int f = 0; f < 2; ++f) {
      FragA pf;
      const unsigned short* pr = &pb[wave][l15 * 64];
      pf.h[0] = *(const u16x8*)(pr + f * 32 + hi * 16);
      pf.h[1] = *(const u16x8*)(pr + f * 32 + hi * 16 + 8);
#pragma unroll
      for (int t = 0; t < 4; ++t) {
        FragB vf;
        vf.u = *(const u16x16*)(vt + (t * 16 + l15) * 64 + f * 32 + hi * 16);
        o[t] = wmma_bf16(pf.v, vf.v, o[t]);
      }
    }
  }

  // Epilogue: O / l  -> attn output [b, n, h*64 + d] (bf16)
#pragma unroll
  for (int r = 0; r < 8; ++r) {
    const float inv = 1.0f / l_i[r];
    const int n = q0 + r + hi * 8;
#pragma unroll
    for (int t = 0; t < 4; ++t) {
      og[((size_t)b * SEQ + n) * DIM + h * DH + t * 16 + l15] =
          f2bf(o[t][r] * inv);
    }
  }
}

// ---------------------------------------------------------------------------
// Output projection: y = attn @ w_out^T + b_out  (M=8192, N=512, K=512), f32 out
// ---------------------------------------------------------------------------
__global__ void __launch_bounds__(128)
out_gemm(const unsigned short* __restrict__ ab,
         const unsigned short* __restrict__ wob,
         const float* __restrict__ bo,
         float* __restrict__ out) {
  const int lane = threadIdx.x & 31;
  const int wave = threadIdx.x >> 5;
  const int l15  = lane & 15;
  const int hi   = lane >> 4;
  const int m0   = blockIdx.x * 16;
  const int e0   = blockIdx.y * 256 + wave * 64;

  v8f acc[4] = {};
  const unsigned short* arow = ab + (size_t)(m0 + l15) * DIM;

  for (int kk = 0; kk < DIM; kk += 32) {
    FragA a;
    a.h[0] = *(const u16x8*)(arow + kk + hi * 16);
    a.h[1] = *(const u16x8*)(arow + kk + hi * 16 + 8);
#pragma unroll
    for (int t = 0; t < 4; ++t) {
      FragB bf;
      bf.u = *(const u16x16*)(wob + (size_t)(e0 + t * 16 + l15) * DIM + kk + hi * 16);
      acc[t] = wmma_bf16(a.v, bf.v, acc[t]);
    }
  }

#pragma unroll
  for (int t = 0; t < 4; ++t) {
#pragma unroll
    for (int r = 0; r < 8; ++r) {
      const int row = m0 + r + hi * 8;
      const int e   = e0 + t * 16 + l15;
      out[(size_t)row * DIM + e] = acc[t][r] + bo[e];
    }
  }
}

// ---------------------------------------------------------------------------
// Host launcher
// ---------------------------------------------------------------------------
extern "C" void kernel_launch(void* const* d_in, const int* in_sizes, int n_in,
                              void* d_out, int out_size, void* d_ws, size_t ws_size,
                              hipStream_t stream) {
  const float* x    = (const float*)d_in[0];  // [2,4096,512]
  const float* wqkv = (const float*)d_in[1];  // [1536,512]
  const float* wout = (const float*)d_in[2];  // [512,512]
  const float* bout = (const float*)d_in[3];  // [512]
  float* out = (float*)d_out;                 // [2,4096,512] f32
  (void)in_sizes; (void)n_in; (void)out_size; (void)ws_size;

  char* ws = (char*)d_ws;
  size_t off = 0;
  auto carve = [&](size_t elems) -> unsigned short* {
    unsigned short* p = (unsigned short*)(ws + off);
    off += (elems * 2 + 255) & ~(size_t)255;
    return p;
  };
  unsigned short* xb  = carve((size_t)ROWS * DIM);         // x in bf16
  unsigned short* wqb = carve((size_t)E_QKV * DIM);        // w_qkv bf16
  unsigned short* wob = carve((size_t)DIM * DIM);          // w_out bf16
  unsigned short* qb  = carve((size_t)BATCH * HEADS * SEQ * DH);  // Q [b,h,n,d]
  unsigned short* kb  = carve((size_t)BATCH * HEADS * SEQ * DH);  // K [b,h,n,d]
  unsigned short* vtb = carve((size_t)BATCH * HEADS * DH * SEQ);  // V^T [b,h,d,n]
  unsigned short* ab  = carve((size_t)ROWS * DIM);         // attn out bf16

  {
    const int n1 = ROWS * DIM;
    cvt_bf16<<<(n1 + 255) / 256, 256, 0, stream>>>(x, xb, n1);
    const int n2 = E_QKV * DIM;
    cvt_bf16<<<(n2 + 255) / 256, 256, 0, stream>>>(wqkv, wqb, n2);
    const int n3 = DIM * DIM;
    cvt_bf16<<<(n3 + 255) / 256, 256, 0, stream>>>(wout, wob, n3);
  }

  qkv_gemm  <<<dim3(ROWS / 16, E_QKV / 256), 128, 0, stream>>>(xb, wqb, qb, kb, vtb);
  flash_attn<<<dim3(SEQ / 64, HEADS, BATCH), 128, 0, stream>>>(qb, kb, vtb, ab);
  out_gemm  <<<dim3(ROWS / 16, DIM / 256),  128, 0, stream>>>(ab, wob, bout, out);
}
